// MoE_4956392259747
// MI455X (gfx1250) — matrix operand order repaired and checked by reading, not compiled, not run on hardware
//
#include <hip/hip_runtime.h>
#include <math.h>

typedef __bf16 bf16;
typedef __attribute__((ext_vector_type(8)))  bf16  bf16x8;
typedef __attribute__((ext_vector_type(16))) bf16  bf16x16;
typedef __attribute__((ext_vector_type(8)))  float f32x8;

#define N_TOK 16384
#define HD    512
#define FD    2048
#define ED    8
#define KSEL  2
#define CAP   5120
#define NK    (N_TOK * KSEL)   // 32768
#define NBLK  (NK / 256)       // 128

#define BM 128
#define BN 128
#define BK 64

__device__ __forceinline__ float gelu_tanh(float v) {
    const float c = 0.7978845608028654f;
    return 0.5f * v * (1.0f + tanhf(c * (v + 0.044715f * v * v * v)));
}

// LDS byte offset of a __shared__ pointer (generic AS: low 32 bits = LDS offset)
__device__ __forceinline__ unsigned lds_u32(const void* p) {
    return (unsigned)(uintptr_t)p;
}
// CDNA5 async global->LDS copy, 16B per lane, tracked by ASYNCcnt
__device__ __forceinline__ void async_b128(unsigned lds_off, const void* g) {
    asm volatile("global_load_async_to_lds_b128 %0, %1, off"
                 :: "v"(lds_off), "v"(g) : "memory");
}
__device__ __forceinline__ void wait_async0() {
    asm volatile("s_wait_asynccnt 0x0" ::: "memory");
}

// ---------------------------------------------------------------------------
// Router: logits = x @ rw, top-2 (jax tie-break: lower index wins), softmax.
// ---------------------------------------------------------------------------
__global__ __launch_bounds__(256) void k_router(const float* __restrict__ x,
                                                const float* __restrict__ rw,
                                                int* __restrict__ eidx,
                                                float* __restrict__ wts) {
    __shared__ float srw[HD * ED];
    for (int i = threadIdx.x; i < HD * ED; i += 256) srw[i] = rw[i];
    __syncthreads();

    int wave = threadIdx.x >> 5, lane = threadIdx.x & 31;
    int t = blockIdx.x * 8 + wave;
    const float* xr = x + (size_t)t * HD;

    float acc[ED];
#pragma unroll
    for (int e = 0; e < ED; ++e) acc[e] = 0.f;

#pragma unroll 4
    for (int i = 0; i < HD / 32; ++i) {
        int h = lane + 32 * i;
        float xv = xr[h];
        const float* w = &srw[h * ED];
#pragma unroll
        for (int e = 0; e < ED; ++e) acc[e] = fmaf(xv, w[e], acc[e]);
    }
#pragma unroll
    for (int off = 16; off > 0; off >>= 1)
#pragma unroll
        for (int e = 0; e < ED; ++e) acc[e] += __shfl_xor(acc[e], off, 32);

    if (lane == 0) {
        int i1 = 0; float v1 = acc[0];
#pragma unroll
        for (int e = 1; e < ED; ++e) if (acc[e] > v1) { v1 = acc[e]; i1 = e; }
        int i2 = -1; float v2 = -3.4e38f;
#pragma unroll
        for (int e = 0; e < ED; ++e) if (e != i1 && acc[e] > v2) { v2 = acc[e]; i2 = e; }
        float e2 = __expf(v2 - v1);          // v1 >= v2
        float z = 1.0f + e2;
        eidx[t * 2 + 0] = i1;  eidx[t * 2 + 1] = i2;
        wts[t * 2 + 0] = 1.0f / z;  wts[t * 2 + 1] = e2 / z;
    }
}

// ---------------------------------------------------------------------------
// Stable rank-within-expert over flat (token,k) order == reference `slot`.
// ---------------------------------------------------------------------------
__global__ __launch_bounds__(256) void k_count(const int* __restrict__ eidx,
                                               int* __restrict__ bc) {
    __shared__ int c[ED];
    if (threadIdx.x < ED) c[threadIdx.x] = 0;
    __syncthreads();
    int i = blockIdx.x * 256 + threadIdx.x;
    atomicAdd(&c[eidx[i]], 1);
    __syncthreads();
    if (threadIdx.x < ED) bc[blockIdx.x * ED + threadIdx.x] = c[threadIdx.x];
}

__global__ void k_scan(const int* __restrict__ bc, int* __restrict__ boff) {
    int e = threadIdx.x;
    if (e < ED) {
        int off = 0;
        for (int b = 0; b < NBLK; ++b) { boff[b * ED + e] = off; off += bc[b * ED + e]; }
    }
}

__global__ __launch_bounds__(256) void k_rank(const int* __restrict__ eidx,
                                              const int* __restrict__ boff,
                                              int* __restrict__ slot) {
    __shared__ int wcnt[8][ED];
    int tid = threadIdx.x, wave = tid >> 5, lane = tid & 31;
    int i = blockIdx.x * 256 + tid;
    int e = eidx[i];
    unsigned masks[ED];
#pragma unroll
    for (int q = 0; q < ED; ++q) masks[q] = (unsigned)__ballot(e == q);
    if (lane < ED) wcnt[wave][lane] = __popc(masks[lane]);
    int inwave = __popc(masks[e] & ((1u << lane) - 1u));
    __syncthreads();
    int pre = 0;
    for (int w = 0; w < wave; ++w) pre += wcnt[w][e];
    slot[i] = boff[blockIdx.x * ED + e] + pre + inwave;
}

// ---------------------------------------------------------------------------
// Pack kept assignments into bf16 expert buffers. One wave per assignment.
// ---------------------------------------------------------------------------
__global__ __launch_bounds__(256) void k_pack(const float* __restrict__ x,
                                              const int* __restrict__ eidx,
                                              const int* __restrict__ slot,
                                              bf16* __restrict__ Xe) {
    int i = blockIdx.x * 8 + (threadIdx.x >> 5);
    int lane = threadIdx.x & 31;
    int s = slot[i];
    if (s >= CAP) return;
    int e = eidx[i];
    int t = i >> 1;
    const float* src = x + (size_t)t * HD;
    bf16* dst = Xe + ((size_t)e * CAP + s) * HD;
#pragma unroll 4
    for (int j = lane; j < HD; j += 32) dst[j] = (bf16)src[j];
}

// ---------------------------------------------------------------------------
// Transpose [E][R][C] f32 -> [E][C][R] bf16 through LDS tiles.
// ---------------------------------------------------------------------------
__global__ __launch_bounds__(256) void k_transpose_bf16(const float* __restrict__ in,
                                                        bf16* __restrict__ out,
                                                        int R, int C) {
    __shared__ float tile[32][33];
    int e = blockIdx.z;
    int r0 = blockIdx.x * 32, c0 = blockIdx.y * 32;
    int tx = threadIdx.x & 31, ty = threadIdx.x >> 5;
    const float* ip = in + (size_t)e * R * C;
    bf16* op = out + (size_t)e * C * R;
#pragma unroll
    for (int p = 0; p < 4; ++p) {
        int r = ty + p * 8;
        tile[r][tx] = ip[(size_t)(r0 + r) * C + c0 + tx];
    }
    __syncthreads();
#pragma unroll
    for (int p = 0; p < 4; ++p) {
        int c = ty + p * 8;
        op[(size_t)(c0 + c) * R + r0 + tx] = (bf16)tile[tx][c];
    }
}

// ---------------------------------------------------------------------------
// Tiled bf16 WMMA GEMM with async double-buffered LDS staging.
//   A:  [E][M][Kd]  (bf16, row-major)
//   Bt: [E][N][Kd]  (bf16, N-major so B fragments pack like A fragments)
// Block tile 128x128x64, 8 waves (2x4), each wave 4x2 accum tiles,
// 16 x v_wmma_f32_16x16x32_bf16 per K-iteration.
// ---------------------------------------------------------------------------
__global__ __launch_bounds__(256) void k_gemm(const bf16* __restrict__ A,
                                              const bf16* __restrict__ Bt,
                                              const float* __restrict__ bias,
                                              bf16* __restrict__ outBf,
                                              float* __restrict__ outF,
                                              int M, int N, int Kd, int doGelu) {
    __shared__ __align__(16) bf16 As[2][BM][BK];
    __shared__ __align__(16) bf16 Bs[2][BN][BK];

    int e = blockIdx.z;
    int m0 = blockIdx.x * BM, n0 = blockIdx.y * BN;
    const bf16* Ae = A  + (size_t)e * M * Kd;
    const bf16* Be = Bt + (size_t)e * N * Kd;

    int tid = threadIdx.x, lane = tid & 31, wave = tid >> 5;
    int wm = wave >> 2, wn = wave & 3;       // 2 x 4 wave grid
    int half = lane >> 4, r = lane & 15;

    f32x8 acc[4][2];
#pragma unroll
    for (int mi = 0; mi < 4; ++mi)
#pragma unroll
        for (int ni = 0; ni < 2; ++ni)
            acc[mi][ni] = (f32x8){0.f, 0.f, 0.f, 0.f, 0.f, 0.f, 0.f, 0.f};

    int srow = tid >> 3;            // 0..31 (row within 32-row pass)
    int scol = (tid & 7) * 8;       // 0..56 (8-elem chunk in 64-wide tile)

    // issue one tile-pair of async global->LDS copies (16KB A + 16KB B)
    auto stage = [&](int k0, int buf) {
#pragma unroll
        for (int p = 0; p < 4; ++p) {
            int row = p * 32 + srow;
            async_b128(lds_u32(&As[buf][row][scol]),
                       &Ae[(size_t)(m0 + row) * Kd + k0 + scol]);
            async_b128(lds_u32(&Bs[buf][row][scol]),
                       &Be[(size_t)(n0 + row) * Kd + k0 + scol]);
        }
    };

    int nIter = Kd / BK;
    stage(0, 0);

    for (int it = 0; it < nIter; ++it) {
        int cur = it & 1;
        wait_async0();          // our async writes into As/Bs[cur] have landed
        __syncthreads();        // everyone's writes landed; prev reads finished
        if (it + 1 < nIter) {
            stage((it + 1) * BK, cur ^ 1);
            __builtin_prefetch(&Ae[(size_t)(m0 + srow) * Kd + (it + 1) * BK], 0, 3);
            __builtin_prefetch(&Be[(size_t)(n0 + srow) * Kd + (it + 1) * BK], 0, 3);
        }

#pragma unroll
        for (int kk = 0; kk < BK / 32; ++kk) {
            bf16x16 af[4], bfr[2];
            // A fragment: half 0 -> K[0..8)+K[16..24), half 1 -> K[8..16)+K[24..32)
#pragma unroll
            for (int mi = 0; mi < 4; ++mi) {
                const bf16* p = &As[cur][wm * 64 + mi * 16 + r][kk * 32];
                bf16x8 lo = *(const bf16x8*)(p + half * 8);
                bf16x8 hi = *(const bf16x8*)(p + 16 + half * 8);
                af[mi] = __builtin_shufflevector(lo, hi,
                         0,1,2,3,4,5,6,7,8,9,10,11,12,13,14,15);
            }
            // B fragment: lane = column, half 0 -> K[0..8)+K[16..24), half 1 -> K[8..16)+K[24..32)
#pragma unroll
            for (int ni = 0; ni < 2; ++ni) {
                const bf16* p = &Bs[cur][wn * 32 + ni * 16 + r][kk * 32];
                bf16x8 lo = *(const bf16x8*)(p + half * 8);
                bf16x8 hi = *(const bf16x8*)(p + 16 + half * 8);
                bfr[ni] = __builtin_shufflevector(lo, hi,
                          0,1,2,3,4,5,6,7,8,9,10,11,12,13,14,15);
            }
#pragma unroll
            for (int mi = 0; mi < 4; ++mi)
#pragma unroll
                for (int ni = 0; ni < 2; ++ni)
                    acc[mi][ni] = __builtin_amdgcn_wmma_f32_16x16x32_bf16(
                        false, af[mi], false, bfr[ni], (short)0, acc[mi][ni],
                        false, false);
        }
        __syncthreads();        // done reading As/Bs[cur]; next iter may refill it
    }

    // Epilogue. C layout: VGPR v, lanes 0-15 -> M=v, lanes 16-31 -> M=v+8; N=r.
#pragma unroll
    for (int mi = 0; mi < 4; ++mi) {
#pragma unroll
        for (int ni = 0; ni < 2; ++ni) {
            int Mb = m0 + wm * 64 + mi * 16 + half * 8;
            int Nc = n0 + wn * 32 + ni * 16 + r;
            float bv = bias[(size_t)e * N + Nc];
#pragma unroll
            for (int v = 0; v < 8; ++v) {
                float val = acc[mi][ni][v] + bv;
                size_t oidx = ((size_t)e * M + Mb + v) * N + Nc;
                if (doGelu) outBf[oidx] = (bf16)gelu_tanh(val);
                else        outF[oidx]  = val;
            }
        }
    }
}

// ---------------------------------------------------------------------------
// Combine: out[t] = sum_k kept ? w_k * out_e[e_k][slot_k] : 0. One wave/token.
// ---------------------------------------------------------------------------
__global__ __launch_bounds__(256) void k_combine(const float* __restrict__ oute,
                                                 const int* __restrict__ eidx,
                                                 const int* __restrict__ slot,
                                                 const float* __restrict__ wts,
                                                 float* __restrict__ out) {
    int t = blockIdx.x * 8 + (threadIdx.x >> 5);
    int lane = threadIdx.x & 31;
    int i0 = t * 2, i1 = t * 2 + 1;
    int s0 = slot[i0], s1 = slot[i1];
    int e0 = eidx[i0], e1 = eidx[i1];
    bool k0 = s0 < CAP, k1 = s1 < CAP;
    const float* p0 = oute + ((size_t)e0 * CAP + (k0 ? s0 : 0)) * HD;
    const float* p1 = oute + ((size_t)e1 * CAP + (k1 ? s1 : 0)) * HD;
    float a0 = k0 ? wts[i0] : 0.f;
    float a1 = k1 ? wts[i1] : 0.f;
    float* dst = out + (size_t)t * HD;
#pragma unroll 4
    for (int j = lane; j < HD; j += 32) {
        float v = 0.f;
        if (k0) v = fmaf(a0, p0[j], v);
        if (k1) v = fmaf(a1, p1[j], v);
        dst[j] = v;
    }
}

// ---------------------------------------------------------------------------
extern "C" void kernel_launch(void* const* d_in, const int* in_sizes, int n_in,
                              void* d_out, int out_size, void* d_ws, size_t ws_size,
                              hipStream_t stream) {
    const float* x   = (const float*)d_in[0];  // [B,T,H] = [n,512]
    const float* rw  = (const float*)d_in[1];  // [512,8]
    const float* w1  = (const float*)d_in[2];  // [8,512,2048]
    const float* b1  = (const float*)d_in[3];  // [8,2048]
    const float* w2  = (const float*)d_in[4];  // [8,2048,512]
    const float* b2  = (const float*)d_in[5];  // [8,512]
    float* out = (float*)d_out;

    size_t off = 0;
    auto nextp = [&](size_t bytes) {
        char* p = (char*)d_ws + off;
        off += (bytes + 255) & ~(size_t)255;
        return (void*)p;
    };
    int*   eidx = (int*)  nextp((size_t)NK * 4);
    float* wts  = (float*)nextp((size_t)NK * 4);
    int*   slot = (int*)  nextp((size_t)NK * 4);
    int*   bc   = (int*)  nextp((size_t)NBLK * ED * 4);
    int*   boff = (int*)  nextp((size_t)NBLK * ED * 4);
    bf16*  Xe   = (bf16*) nextp((size_t)ED * CAP * HD * 2);
    bf16*  w1t  = (bf16*) nextp((size_t)ED * FD * HD * 2);   // [E][F][H]
    bf16*  w2t  = (bf16*) nextp((size_t)ED * HD * FD * 2);   // [E][H][F]
    bf16*  hbuf = (bf16*) nextp((size_t)ED * CAP * FD * 2);  // [E][cap][F]
    float* oute = (float*)nextp((size_t)ED * CAP * HD * 4);  // [E][cap][H]

    // Weight convert+transpose to bf16 N-major
    k_transpose_bf16<<<dim3(HD / 32, FD / 32, ED), 256, 0, stream>>>(w1, w1t, HD, FD);
    k_transpose_bf16<<<dim3(FD / 32, HD / 32, ED), 256, 0, stream>>>(w2, w2t, FD, HD);

    // Router + top-2 + softmax
    k_router<<<N_TOK / 8, 256, 0, stream>>>(x, rw, eidx, wts);

    // Stable slot ranks
    k_count<<<NBLK, 256, 0, stream>>>(eidx, bc);
    k_scan<<<1, 32, 0, stream>>>(bc, boff);
    k_rank<<<NBLK, 256, 0, stream>>>(eidx, boff, slot);

    // Pack expert inputs (bf16)
    k_pack<<<NK / 8, 256, 0, stream>>>(x, eidx, slot, Xe);

    // GEMM1: h = gelu(Xe @ w1 + b1)   [E][cap=5120][F=2048], K=512
    k_gemm<<<dim3(CAP / BM, FD / BN, ED), 256, 0, stream>>>(
        Xe, w1t, b1, hbuf, nullptr, CAP, FD, HD, 1);

    // GEMM2: oute = h @ w2 + b2       [E][cap][H=512], K=2048
    k_gemm<<<dim3(CAP / BM, HD / BN, ED), 256, 0, stream>>>(
        hbuf, w2t, b2, nullptr, oute, CAP, HD, FD, 0);

    // Weighted scatter-combine (deterministic, no atomics)
    k_combine<<<N_TOK / 8, 256, 0, stream>>>(oute, eidx, slot, wts, out);

    (void)in_sizes; (void)n_in; (void)out_size; (void)ws_size;
}
